// TreeLSTM_82403242541826
// MI455X (gfx1250) — matrix operand order repaired and checked by reading, not compiled, not run on hardware
//
#include <hip/hip_runtime.h>

// TreeLSTM for MI455X (gfx1250): bf16 WMMA GEMMs + async Global->LDS A-staging
// + fused activation epilogues.  B=128, DEPTH=10, H=256, E=256, NTREE=1023.
//
// d_in: 0 wordid(int32 B*1023), 1 mask(unused), 2 embedding(f32 30000*256),
//       3 W_iou(f32 768*256), 4 U_iou(f32 768*512), 5 b_iou(f32 768),
//       6 U_f_w(f32 512*512), 7 U_f_b(f32 512)
// d_out: h (B*1023*256 f32) then c (B*1023*256 f32)
// d_ws:  bf16 W_iou copy (768*256) + bf16 [U_f_w;U_iou] (1280*512)

#define BNH ((size_t)128 * 1023 * 256)

#if defined(__HIP_DEVICE_COMPILE__)

#if __has_builtin(__builtin_amdgcn_global_load_async_to_lds_b128) && \
    __has_builtin(__builtin_amdgcn_s_wait_asynccnt)
#define USE_ASYNC 1
#else
#define USE_ASYNC 0
#endif
#if !__has_builtin(__builtin_amdgcn_global_load_async_to_lds_b128)
#warning "probe: NO __builtin_amdgcn_global_load_async_to_lds_b128"
#endif
#if !__has_builtin(__builtin_amdgcn_s_wait_asynccnt)
#warning "probe: NO __builtin_amdgcn_s_wait_asynccnt"
#endif

typedef __bf16 v16bf __attribute__((ext_vector_type(16)));
typedef float  v8f   __attribute__((ext_vector_type(8)));
typedef int    v4i_  __attribute__((vector_size(4 * sizeof(int))));

typedef __attribute__((address_space(1))) v4i_* gv4i_p;
typedef __attribute__((address_space(3))) v4i_* lv4i_p;

__device__ __forceinline__ float sigm(float x) { return 1.0f / (1.0f + __expf(-x)); }

// Move one 16B chunk global->LDS (async path, tracked by ASYNCcnt).
__device__ __forceinline__ void stage16(const float* g, float* l) {
#if USE_ASYNC
  __builtin_amdgcn_global_load_async_to_lds_b128((gv4i_p)g, (lv4i_p)l, 0, 0);
#else
  *(float4*)l = *(const float4*)g;
#endif
}
__device__ __forceinline__ void stage_fence() {
#if USE_ASYNC
  __builtin_amdgcn_s_wait_asynccnt(0);
#endif
  __syncthreads();
}

// A-fragment (16x32 bf16) built from an f32 row in LDS:
// lane half h: elements 0..15 = K k+16h..k+16h+15 (contiguous, matches B)
__device__ __forceinline__ v16bf load_a_lds(const float* ap, int half) {
  v16bf a;
  const float* c0 = ap + half * 16;
  const float* c1 = ap + half * 16 + 8;
#pragma unroll
  for (int i = 0; i < 8; ++i) {
    a[i]     = (__bf16)c0[i];
    a[i + 8] = (__bf16)c1[i];
  }
  return a;
}
#endif  // __HIP_DEVICE_COMPILE__

// ---------------------------------------------------------------- weights
__global__ __launch_bounds__(256) void wconv_kernel(const float* __restrict__ W_iou,
                                                    const float* __restrict__ U_iou,
                                                    const float* __restrict__ U_f_w,
                                                    unsigned short* __restrict__ Wleaf_us,
                                                    unsigned short* __restrict__ Ucomb_us) {
#if defined(__HIP_DEVICE_COMPILE__)
  __bf16* Wleaf = (__bf16*)Wleaf_us;
  __bf16* Ucomb = (__bf16*)Ucomb_us;
  int stride = gridDim.x * blockDim.x;
  int tid = blockIdx.x * blockDim.x + threadIdx.x;
  for (int i = tid; i < 768 * 256; i += stride) Wleaf[i] = (__bf16)W_iou[i];
  for (int i = tid; i < 1280 * 512; i += stride) {
    int n = i >> 9, k = i & 511;
    float v = (n < 512) ? U_f_w[n * 512 + k] : U_iou[(n - 512) * 512 + k];
    Ucomb[i] = (__bf16)v;
  }
#endif
}

// ---------------------------------------------------------------- leaves
// rows m = b*512 + li (node = 511+li), M-block = 16 rows, N = 768, K = 256.
// A tile (16x256 f32 = 16 KB) staged into LDS async; 8 waves x 6 N-tiles.
__global__ __launch_bounds__(256) void leaf_kernel(const int* __restrict__ wordid,
                                                   const float* __restrict__ emb,
                                                   const unsigned short* __restrict__ Wleaf_us,
                                                   const float* __restrict__ b_iou,
                                                   float* __restrict__ hptr,
                                                   float* __restrict__ cptr) {
#if defined(__HIP_DEVICE_COMPILE__)
  __shared__ float smem[16 * 768];  // [0..4095]: A tile; later 16x768 iou scratch
  const __bf16* Wleaf = (const __bf16*)Wleaf_us;
  const int tid = threadIdx.x;
  const int wave = tid >> 5, lane = tid & 31;
  const int half = lane >> 4, r = lane & 15;
  const int mbase = blockIdx.x * 16;

  // ---- stage A: 1024 x 16B chunks, 4 per thread (embedding row gather)
#pragma unroll
  for (int i = 0; i < 4; ++i) {
    int c = tid + i * 256;        // 0..1023
    int row = c >> 6;             // 16 rows
    int col = (c & 63) * 4;       // 0..252
    int mm = mbase + row;
    int bb = mm >> 9;
    int node = 511 + (mm & 511);
    int word = wordid[bb * 1023 + node];
    stage16(emb + (size_t)word * 256 + col, &smem[row * 256 + col]);
  }
  stage_fence();

  v8f acc[6];
#pragma unroll
  for (int t = 0; t < 6; ++t)
#pragma unroll
    for (int v = 0; v < 8; ++v) acc[t][v] = 0.0f;

  const __bf16* brow = Wleaf + (size_t)(wave * 96 + r) * 256;

  for (int k = 0; k < 256; k += 32) {
    v16bf a = load_a_lds(&smem[r * 256 + k], half);
#pragma unroll
    for (int t = 0; t < 6; ++t) {
      v16bf bf = *(const v16bf*)(brow + (size_t)t * 16 * 256 + k + half * 16);
      acc[t] = __builtin_amdgcn_wmma_f32_16x16x32_bf16(false, a, false, bf, (short)0,
                                                       acc[t], false, false);
    }
  }
  __syncthreads();  // all waves done reading A tile before overwriting smem

#pragma unroll
  for (int t = 0; t < 6; ++t)
#pragma unroll
    for (int v = 0; v < 8; ++v)
      smem[(v + 8 * half) * 768 + wave * 96 + 16 * t + r] = acc[t][v];
  __syncthreads();

  const int j = tid;  // 0..255
  for (int rr = 0; rr < 16; ++rr) {
    const float* row = smem + rr * 768;
    float ii = sigm(row[j] + b_iou[j]);
    float oo = sigm(row[256 + j] + b_iou[256 + j]);
    float uu = tanhf(row[512 + j] + b_iou[512 + j]);
    float cc = ii * uu;
    float hh = oo * tanhf(cc);
    int mm = mbase + rr;
    int bb = mm >> 9;
    int nn = 511 + (mm & 511);
    size_t off = ((size_t)(bb * 1023 + nn)) * 256 + j;
    hptr[off] = hh;
    cptr[off] = cc;
  }
#endif
}

// ---------------------------------------------------------------- internal levels
// rows m = b*P + pi (node = P-1+pi), A = [h(child0), h(child1)] (K=512) staged
// into LDS (16x512 f32 = 32 KB) async.  N = 1280 = [512 f-raw | 768 iou].
// Waves 0..3 own f cols {w*64..} U {256+w*64..} and fold c_red into LDS;
// waves 4..7 own iou cols.  LDS 16x1024 f32: cols 0..767 iou, 768..1023 c_red.
__global__ __launch_bounds__(256) void level_kernel(const unsigned short* __restrict__ Ucomb_us,
                                                    const float* __restrict__ U_f_b,
                                                    const float* __restrict__ b_iou,
                                                    float* __restrict__ hptr,
                                                    float* __restrict__ cptr,
                                                    int lvl) {
#if defined(__HIP_DEVICE_COMPILE__)
  __shared__ float smem[16 * 1024];  // [0..8191]: A tile; later 16x1024 scratch
  const __bf16* Ucomb = (const __bf16*)Ucomb_us;
  const int P = 1 << lvl;
  const int tid = threadIdx.x;
  const int wave = tid >> 5, lane = tid & 31;
  const int half = lane >> 4, r = lane & 15;
  const int mbase = blockIdx.x * 16;

  // ---- stage A: h_cat rows, 2048 x 16B chunks, 8 per thread
#pragma unroll
  for (int i = 0; i < 8; ++i) {
    int c = tid + i * 256;        // 0..2047
    int row = c >> 7;             // 16 rows
    int col = (c & 127) * 4;      // 0..508
    int mm = mbase + row;
    int bb = mm >> lvl;
    int qq = (P - 1) + (mm & (P - 1));
    const float* g =
        hptr + ((size_t)(bb * 1023 + 2 * qq + 1 + (col >= 256))) * 256 + (col & 255);
    stage16(g, &smem[row * 512 + col]);
  }

  // ---- prefetch child-c rows needed by the f-gate epilogue (32 KB / block)
  if (wave < 4) {
    int idx = wave * 32 + lane;   // 0..127
    int prow = idx >> 3;          // 0..15
    int sub = idx & 7;            // 0..7 : child = sub>>2, quarter = sub&3
    int mm = mbase + prow;
    int bb = mm >> lvl;
    int qq = (P - 1) + (mm & (P - 1));
    const float* cr =
        cptr + ((size_t)(bb * 1023 + 2 * qq + 1 + (sub >> 2))) * 256 + (sub & 3) * 64;
    __builtin_prefetch(cr, 0, 0);
  }
  stage_fence();

  if (wave < 4) {
    // f part: tiles 0..3 -> cols wave*64+16t ; tiles 4..7 -> 256+wave*64+16t
    v8f acc[8];
#pragma unroll
    for (int t = 0; t < 8; ++t)
#pragma unroll
      for (int v = 0; v < 8; ++v) acc[t][v] = 0.0f;

    const __bf16* brow0 = Ucomb + (size_t)(wave * 64 + r) * 512;
    const __bf16* brow1 = Ucomb + (size_t)(256 + wave * 64 + r) * 512;

    for (int k = 0; k < 512; k += 32) {
      v16bf a = load_a_lds(&smem[r * 512 + k], half);
#pragma unroll
      for (int t = 0; t < 4; ++t) {
        v16bf bf0 = *(const v16bf*)(brow0 + (size_t)t * 16 * 512 + k + half * 16);
        acc[t] = __builtin_amdgcn_wmma_f32_16x16x32_bf16(false, a, false, bf0, (short)0,
                                                         acc[t], false, false);
        v16bf bf1 = *(const v16bf*)(brow1 + (size_t)t * 16 * 512 + k + half * 16);
        acc[4 + t] = __builtin_amdgcn_wmma_f32_16x16x32_bf16(false, a, false, bf1, (short)0,
                                                             acc[4 + t], false, false);
      }
    }
    __syncthreads();  // all waves done reading A tile
    // c_red = sig(f0)*c_child0 + sig(f1)*c_child1 -> LDS cols 768..1023
#pragma unroll
    for (int t = 0; t < 4; ++t)
#pragma unroll
      for (int v = 0; v < 8; ++v) {
        int j = wave * 64 + 16 * t + r;
        float f0 = sigm(acc[t][v] + U_f_b[j]);
        float f1 = sigm(acc[4 + t][v] + U_f_b[256 + j]);
        int ml = v + 8 * half;
        int mm = mbase + ml;
        int bb = mm >> lvl;
        int qq = (P - 1) + (mm & (P - 1));
        float c0v = cptr[((size_t)(bb * 1023 + 2 * qq + 1)) * 256 + j];
        float c1v = cptr[((size_t)(bb * 1023 + 2 * qq + 2)) * 256 + j];
        smem[ml * 1024 + 768 + j] = f0 * c0v + f1 * c1v;
      }
  } else {
    // iou part: 12 tiles, cols 512 + (wave-4)*192 + 16t (LDS col = n-512)
    v8f acc[12];
#pragma unroll
    for (int t = 0; t < 12; ++t)
#pragma unroll
      for (int v = 0; v < 8; ++v) acc[t][v] = 0.0f;

    const __bf16* brow = Ucomb + (size_t)(512 + (wave - 4) * 192 + r) * 512;

    for (int k = 0; k < 512; k += 32) {
      v16bf a = load_a_lds(&smem[r * 512 + k], half);
#pragma unroll
      for (int t = 0; t < 12; ++t) {
        v16bf bf = *(const v16bf*)(brow + (size_t)t * 16 * 512 + k + half * 16);
        acc[t] = __builtin_amdgcn_wmma_f32_16x16x32_bf16(false, a, false, bf, (short)0,
                                                         acc[t], false, false);
      }
    }
    __syncthreads();  // all waves done reading A tile
#pragma unroll
    for (int t = 0; t < 12; ++t)
#pragma unroll
      for (int v = 0; v < 8; ++v)
        smem[(v + 8 * half) * 1024 + (wave - 4) * 192 + 16 * t + r] = acc[t][v];
  }
  __syncthreads();

  const int j = tid;  // 0..255
  for (int rr = 0; rr < 16; ++rr) {
    const float* row = smem + rr * 1024;
    float ii = sigm(row[j] + b_iou[j]);
    float oo = sigm(row[256 + j] + b_iou[256 + j]);
    float uu = tanhf(row[512 + j] + b_iou[512 + j]);
    float cc = ii * uu + row[768 + j];
    float hh = oo * tanhf(cc);
    int mm = mbase + rr;
    int bb = mm >> lvl;
    int qq = (P - 1) + (mm & (P - 1));
    size_t off = ((size_t)(bb * 1023 + qq)) * 256 + j;
    hptr[off] = hh;
    cptr[off] = cc;
  }
#endif
}

// ---------------------------------------------------------------- host
extern "C" void kernel_launch(void* const* d_in, const int* in_sizes, int n_in,
                              void* d_out, int out_size, void* d_ws, size_t ws_size,
                              hipStream_t stream) {
  const int*   wordid = (const int*)d_in[0];
  // d_in[1] (mask) unused: mask==1 on all leaves, internal nodes overwritten.
  const float* emb    = (const float*)d_in[2];
  const float* W_iou  = (const float*)d_in[3];
  const float* U_iou  = (const float*)d_in[4];
  const float* b_iou  = (const float*)d_in[5];
  const float* U_f_w  = (const float*)d_in[6];
  const float* U_f_b  = (const float*)d_in[7];

  float* hptr = (float*)d_out;
  float* cptr = hptr + BNH;

  unsigned short* Wleaf = (unsigned short*)d_ws;      // 768*256 bf16
  unsigned short* Ucomb = Wleaf + (size_t)768 * 256;  // 1280*512 bf16

  wconv_kernel<<<512, 256, 0, stream>>>(W_iou, U_iou, U_f_w, Wleaf, Ucomb);

  // leaves: B*512 rows / 16 per block
  leaf_kernel<<<4096, 256, 0, stream>>>(wordid, emb, Wleaf, b_iou, hptr, cptr);

  // internal levels, bottom-up
  for (int lvl = 8; lvl >= 0; --lvl) {
    int blocks = (128 << lvl) / 16;  // = 8 << lvl
    level_kernel<<<blocks, 256, 0, stream>>>(Ucomb, U_f_b, b_iou, hptr, cptr, lvl);
  }
}
